// MoEFFN_17334488007373
// MI455X (gfx1250) — compile-verified, matrix operand order checked by reading, not run on hardware
//
#include <hip/hip_runtime.h>
#include <hip/hip_bf16.h>
#include <math.h>

#define D_MODEL 768
#define D_HID   3072
#define NEXP    8
#define NTOK    2048
#define TILE_M  16
#define HC      128                     // H-chunk per double-GEMM stage
#define HSPLIT  4                       // H-dimension split across workgroups
#define HSLICE  (D_HID / HSPLIT)        // 768 H per workgroup
#define TILES_PER_E (NTOK / TILE_M)     // 128

typedef __bf16 bf16_t;
typedef bf16_t v16bf __attribute__((ext_vector_type(16)));
typedef float  v8f   __attribute__((ext_vector_type(8)));

union ABfrag { v16bf v; uint4 q[2]; };

__device__ __forceinline__ unsigned short f2bf(float f) {
  union { float f; unsigned int u; } v; v.f = f;
  unsigned int u = v.u;
  u += 0x7FFFu + ((u >> 16) & 1u);   // round-to-nearest-even
  return (unsigned short)(u >> 16);
}

// ---------------------------------------------------------------- zero
__global__ void zero_kernel(float* __restrict__ out, int n, int* __restrict__ counts) {
  int stride = gridDim.x * blockDim.x;
  for (int i = blockIdx.x * blockDim.x + threadIdx.x; i < n; i += stride) out[i] = 0.f;
  if (blockIdx.x == 0 && threadIdx.x < NEXP) counts[threadIdx.x] = 0;
}

// ------------------------------------------- f32 -> bf16 tiled transpose
// src: per-expert [R][C] f32 ; dst: per-expert [C][R] bf16
__global__ void transpose_cvt(const float* __restrict__ src, unsigned short* __restrict__ dst,
                              int R, int C) {
  __shared__ unsigned short tile[32][33];
  const int e = blockIdx.z;
  const float*      s = src + (size_t)e * R * C;
  unsigned short*   d = dst + (size_t)e * R * C;
  const int c0 = blockIdx.x * 32, r0 = blockIdx.y * 32;
  const int tx = threadIdx.x, ty = threadIdx.y;
  #pragma unroll
  for (int i = 0; i < 32; i += 8)
    tile[ty + i][tx] = f2bf(s[(size_t)(r0 + ty + i) * C + (c0 + tx)]);
  __syncthreads();
  #pragma unroll
  for (int i = 0; i < 32; i += 8)
    d[(size_t)(c0 + ty + i) * R + (r0 + tx)] = tile[tx][ty + i];
}

// ---------------------------------------------------------------- x -> bf16
__global__ void cvt_x(const float* __restrict__ src, unsigned short* __restrict__ dst, int n) {
  int i = blockIdx.x * blockDim.x + threadIdx.x;
  if (i < n) dst[i] = f2bf(src[i]);
}

// ---------------------------------------------------------------- router
__global__ void router_kernel(const float* __restrict__ x, const float* __restrict__ gate,
                              int* __restrict__ counts, int* __restrict__ lists,
                              float* __restrict__ wlists) {
  int t = blockIdx.x * blockDim.x + threadIdx.x;
  if (t >= NTOK) return;
  const float* xr = x + (size_t)t * D_MODEL;
  float acc[NEXP];
  #pragma unroll
  for (int e = 0; e < NEXP; e++) acc[e] = 0.f;
  for (int d = 0; d < D_MODEL; d++) {
    float xv = xr[d];
    const float4* g4 = (const float4*)(gate + d * NEXP);
    float4 a = g4[0], b = g4[1];
    acc[0] += xv * a.x; acc[1] += xv * a.y; acc[2] += xv * a.z; acc[3] += xv * a.w;
    acc[4] += xv * b.x; acc[5] += xv * b.y; acc[6] += xv * b.z; acc[7] += xv * b.w;
  }
  float m = acc[0];
  #pragma unroll
  for (int e = 1; e < NEXP; e++) m = fmaxf(m, acc[e]);
  float p[NEXP];
  #pragma unroll
  for (int e = 0; e < NEXP; e++) p[e] = __expf(acc[e] - m);
  int i1 = 0; float p1 = p[0];
  #pragma unroll
  for (int e = 1; e < NEXP; e++) if (p[e] > p1) { p1 = p[e]; i1 = e; }
  int i2 = -1; float p2 = -1.f;
  #pragma unroll
  for (int e = 0; e < NEXP; e++) if (e != i1 && p[e] > p2) { p2 = p[e]; i2 = e; }
  float inv = 1.f / (p1 + p2);
  int pos1 = atomicAdd(&counts[i1], 1);
  lists[i1 * NTOK + pos1] = t;  wlists[i1 * NTOK + pos1] = p1 * inv;
  int pos2 = atomicAdd(&counts[i2], 1);
  lists[i2 * NTOK + pos2] = t;  wlists[i2 * NTOK + pos2] = p2 * inv;
}

// ---------------------------------------------------------------- MoE FFN
// grid: NEXP * HSPLIT * TILES_PER_E blocks of 256 threads (8 wave32s)
// block -> (expert e, H-slice hs, token-tile): workgroups sharing a weight
// slice are adjacent in launch order for L2 temporal locality.
__global__ void __launch_bounds__(256)
moe_ffn_kernel(const unsigned short* __restrict__ xb,    // [N][D] bf16
               const unsigned short* __restrict__ w1b,   // [E][H][D] bf16 (B1 col-major)
               const unsigned short* __restrict__ w2b,   // [E][D][H] bf16 (B2 col-major)
               const float* __restrict__ b1,             // [E][H]
               const float* __restrict__ b2,             // [E][D]
               const int* __restrict__ counts,
               const int* __restrict__ lists,
               const float* __restrict__ wlists,
               float* __restrict__ out) {
  __shared__ __align__(16) unsigned short x_tile[TILE_M * D_MODEL]; // 24 KB
  __shared__ __align__(16) unsigned short h_tile[TILE_M * HC];      // 4 KB
  __shared__ int   tok_row[TILE_M];
  __shared__ float w_row[TILE_M];

  const int tile = blockIdx.x & (TILES_PER_E - 1);
  const int hs   = (blockIdx.x >> 7) & (HSPLIT - 1);
  const int e    = blockIdx.x >> 9;               // 128*4 blocks per expert
  const int cnt  = counts[e];
  const int m0   = tile * TILE_M;
  if (m0 >= cnt) return;                 // uniform exit: EXEC stays all-ones

  const int tid  = threadIdx.x;
  const int lane = tid & 31;
  const int wave = tid >> 5;
  const int lrow = lane & 15;            // M (A) or N (B/C) index
  const int ksel = lane >> 4;            // selects K sub-block of the fragment

  if (tid < TILE_M) {
    int i = m0 + tid;
    bool valid = i < cnt;
    tok_row[tid] = valid ? lists[e * NTOK + i]  : 0;
    w_row[tid]   = valid ? wlists[e * NTOK + i] : 0.f;
  }
  __syncthreads();

  // stage 16 token rows of bf16 x into LDS (zeros for padded rows)
  {
    const int cpr = D_MODEL / 8;                 // 96 16-byte chunks per row
    for (int idx = tid; idx < TILE_M * cpr; idx += 256) {
      int r = idx / cpr, c = idx % cpr;
      uint4 val = make_uint4(0u, 0u, 0u, 0u);
      if (m0 + r < cnt)
        val = *(const uint4*)(xb + (size_t)tok_row[r] * D_MODEL + c * 8);
      *(uint4*)(&x_tile[r * D_MODEL + c * 8]) = val;
    }
  }

  const v8f zero8 = {0.f,0.f,0.f,0.f,0.f,0.f,0.f,0.f};
  v8f yacc[6];                                   // wave owns y cols [wave*96, wave*96+96)
  #pragma unroll
  for (int j = 0; j < 6; j++) yacc[j] = zero8;

  const size_t w1base = (size_t)e * D_HID * D_MODEL;
  const size_t w2base = (size_t)e * D_MODEL * D_HID;
  const int hlo = hs * HSLICE;
  const int hhi = hlo + HSLICE;

  for (int hb = hlo; hb < hhi; hb += HC) {
    __syncthreads();                             // h_tile reusable; x_tile staged

    // ---- GEMM1: h[16][HC]; wave computes its own 16-wide column subtile
    v8f hacc = zero8;
    const int hcol = hb + wave * 16 + lrow;      // this lane's h column
    #pragma unroll 4
    for (int kc = 0; kc < D_MODEL; kc += 32) {
      const int ka = kc + ksel * 8;
      ABfrag a, b;
      a.q[0] = *(const uint4*)(&x_tile[lrow * D_MODEL + ka]);
      a.q[1] = *(const uint4*)(&x_tile[lrow * D_MODEL + ka + 16]);
      const unsigned short* bp = w1b + w1base + (size_t)hcol * D_MODEL + ka;
      b.q[0] = *(const uint4*)(bp);
      b.q[1] = *(const uint4*)(bp + 16);
      hacc = __builtin_amdgcn_wmma_f32_16x16x32_bf16(false, a.v, false, b.v,
                                                     (short)0, hacc, false, false);
    }
    // bias + exact-erf GELU, spill to LDS as bf16 A-operand for GEMM2
    const float b1v = b1[e * D_HID + hcol];
    #pragma unroll
    for (int r = 0; r < 8; r++) {
      float v = hacc[r] + b1v;
      float g = 0.5f * v * (1.f + erff(v * 0.70710678118654752f));
      int m = r + ksel * 8;
      h_tile[m * HC + wave * 16 + lrow] = f2bf(g);
    }
    __syncthreads();                             // h_tile visible to all waves

    // ---- GEMM2: yacc += gelu(h)[16][HC] @ w2[hb:hb+HC][own 96 cols]
    #pragma unroll
    for (int kc = 0; kc < HC; kc += 32) {
      const int ka = kc + ksel * 8;
      ABfrag a;
      a.q[0] = *(const uint4*)(&h_tile[lrow * HC + ka]);
      a.q[1] = *(const uint4*)(&h_tile[lrow * HC + ka + 16]);
      const int kg = hb + ka;
      #pragma unroll
      for (int j = 0; j < 6; j++) {
        const int n = wave * 96 + j * 16 + lrow;
        ABfrag b;
        const unsigned short* bp = w2b + w2base + (size_t)n * D_HID + kg;
        b.q[0] = *(const uint4*)(bp);
        b.q[1] = *(const uint4*)(bp + 16);
        yacc[j] = __builtin_amdgcn_wmma_f32_16x16x32_bf16(false, a.v, false, b.v,
                                                          (short)0, yacc[j], false, false);
      }
    }
  }

  // ---- epilogue: + b2 (slice 0 only), scale by routing weight, atomic combine
  #pragma unroll
  for (int j = 0; j < 6; j++) {
    const int n = wave * 96 + j * 16 + lrow;
    const float b2v = (hs == 0) ? b2[e * D_MODEL + n] : 0.f;
    #pragma unroll
    for (int r = 0; r < 8; r++) {
      const int m = r + ksel * 8;
      const float wgt = w_row[m];
      if (wgt != 0.f)
        atomicAdd(&out[(size_t)tok_row[m] * D_MODEL + n], (yacc[j][r] + b2v) * wgt);
    }
  }
}

// ---------------------------------------------------------------- launcher
extern "C" void kernel_launch(void* const* d_in, const int* in_sizes, int n_in,
                              void* d_out, int out_size, void* d_ws, size_t ws_size,
                              hipStream_t stream) {
  const float* x     = (const float*)d_in[0];   // [1,2048,768]
  const float* gatew = (const float*)d_in[1];   // [768,8]
  const float* w1    = (const float*)d_in[2];   // [8,768,3072]
  const float* b1    = (const float*)d_in[3];   // [8,1,3072]
  const float* w2    = (const float*)d_in[4];   // [8,3072,768]
  const float* b2    = (const float*)d_in[5];   // [8,1,768]
  float* out = (float*)d_out;

  char* ws = (char*)d_ws;
  const size_t wElems = (size_t)NEXP * D_HID * D_MODEL;
  unsigned short* w1b = (unsigned short*)ws;  ws += wElems * 2;                 // 37.7 MB
  unsigned short* w2b = (unsigned short*)ws;  ws += wElems * 2;                 // 37.7 MB
  unsigned short* xb  = (unsigned short*)ws;  ws += (size_t)NTOK * D_MODEL * 2; // 3.1 MB
  int*   counts = (int*)ws;                   ws += NEXP * sizeof(int);
  int*   lists  = (int*)ws;                   ws += (size_t)NEXP * NTOK * sizeof(int);
  float* wlists = (float*)ws;                 ws += (size_t)NEXP * NTOK * sizeof(float);

  zero_kernel<<<256, 256, 0, stream>>>(out, NTOK * D_MODEL, counts);

  dim3 tb(32, 8);
  // w1 [E][D][H] -> w1b [E][H][D]
  transpose_cvt<<<dim3(D_HID / 32, D_MODEL / 32, NEXP), tb, 0, stream>>>(w1, w1b, D_MODEL, D_HID);
  // w2 [E][H][D] -> w2b [E][D][H]
  transpose_cvt<<<dim3(D_MODEL / 32, D_HID / 32, NEXP), tb, 0, stream>>>(w2, w2b, D_HID, D_MODEL);

  cvt_x<<<(NTOK * D_MODEL) / 256, 256, 0, stream>>>(x, xb, NTOK * D_MODEL);

  router_kernel<<<NTOK / 256, 256, 0, stream>>>(x, gatew, counts, lists, wlists);

  moe_ffn_kernel<<<NEXP * HSPLIT * TILES_PER_E, 256, 0, stream>>>(xb, w1b, w2b, b1, b2,
                                                                  counts, lists, wlists, out);
}
